// MultiHeadsAttention_44487271252735
// MI455X (gfx1250) — matrix operand order repaired and checked by reading, not compiled, not run on hardware
//
#include <hip/hip_runtime.h>

#define BATCH 2
#define NSEQ  4096
#define HID_  512
#define HEADS 8
#define DK_   64
#define BN    (BATCH * NSEQ)   // 8192 rows for the projection GEMMs

typedef __attribute__((ext_vector_type(16))) __bf16 v16bf;
typedef __attribute__((ext_vector_type(8)))  float  v8f;

union FragAB {
    v16bf          v;
    unsigned short s[16];
    unsigned       u[8];
};

// Native bf16 convert: backend emits v_cvt_pk_bf16_f32 on gfx1250.
__device__ __forceinline__ unsigned short f2bf(float f) {
    return __builtin_bit_cast(unsigned short, (__bf16)f);
}

// ---------------------------------------------------------------------------
// Tensor Data Mover: 2D bf16 tile load, global -> LDS.  D# per ISA ch.8.
// Optional LDS padding: pad_amount+1 dwords inserted every 2^(pad_interval+1)
// dwords stored (used to de-conflict LDS banks for the K tile).
// ---------------------------------------------------------------------------
#if defined(__HIP_DEVICE_COMPILE__) && __has_builtin(__builtin_amdgcn_tensor_load_to_lds)
#define HAVE_TDM 1
#else
#define HAVE_TDM 0
#endif

#if HAVE_TDM
typedef __attribute__((ext_vector_type(4))) unsigned tdm_v4u;
typedef __attribute__((ext_vector_type(8))) int      tdm_v8i;
typedef __attribute__((ext_vector_type(4))) int      tdm_v4i;

__device__ __forceinline__ void tdm_load_bf16_2d(
    unsigned lds_byte_off,          // LDS byte address of tile destination
    const void* gptr,               // global address of tile start
    unsigned tile_cols,             // tile_dim0 / tensor_dim0 (contiguous, elems)
    unsigned tile_rows,             // tile_dim1 (rows)
    unsigned row_stride,            // tensor_dim0_stride (elems)
    unsigned tensor_rows,           // tensor_dim1 (for OOB)
    unsigned pad_en = 0u,           // 1 = pad LDS destination
    unsigned pad_interval = 0u,     // pad every 2^(k+1) dwords
    unsigned pad_amount = 0u)       // insert (k+1) dwords
{
    unsigned long long ga = (unsigned long long)gptr;
    tdm_v4u g0;
    g0.x = 1u;                                                  // count=1, user
    g0.y = lds_byte_off;                                        // lds_addr
    g0.z = (unsigned)ga;                                        // global[31:0]
    g0.w = (unsigned)((ga >> 32) & 0x01FFFFFFu) | (2u << 30);   // global[56:32], type=2
    tdm_v8i g1;
    g1[0] = (int)((1u << 16) | (pad_en << 20) |
                  (pad_interval << 22) | (pad_amount << 25));   // data_size=1 (2B)
    g1[1] = (int)((tile_cols & 0xFFFFu) << 16);                 // tensor_dim0[15:0]
    g1[2] = (int)(((tile_cols >> 16) & 0xFFFFu) |
                  ((tensor_rows & 0xFFFFu) << 16));             // td0 hi | td1 lo
    g1[3] = (int)(((tensor_rows >> 16) & 0xFFFFu) |
                  ((tile_cols & 0xFFFFu) << 16));               // td1 hi | tile_dim0
    g1[4] = (int)(tile_rows & 0xFFFFu);                         // tile_dim1; tile_dim2=0
    g1[5] = (int)row_stride;                                    // dim0_stride lo32
    g1[6] = 0;
    g1[7] = 0;
    tdm_v4i z4 = {0, 0, 0, 0};                                  // 2D: groups 2/3 unused
#if __clang_major__ >= 23
    tdm_v8i z8 = {0, 0, 0, 0, 0, 0, 0, 0};
    __builtin_amdgcn_tensor_load_to_lds(g0, g1, z4, z4, z8, 0);
#else
    __builtin_amdgcn_tensor_load_to_lds(g0, g1, z4, z4, 0);
#endif
}
#endif

// ---------------------------------------------------------------------------
// Kernel 0: one-shot fp32 -> bf16 weight conversion (512x512).
// ---------------------------------------------------------------------------
__global__ void __launch_bounds__(256)
wcvt_kernel(const float* __restrict__ W, unsigned short* __restrict__ Wb)
{
    int i = blockIdx.x * 256 + threadIdx.x;
    Wb[i] = f2bf(W[i]);
}

// ---------------------------------------------------------------------------
// Kernel 1: Y = X @ W^T + b  (fp32 X, bf16 W, bf16 out in [B,H,N,DK] layout)
// ---------------------------------------------------------------------------
__global__ void __launch_bounds__(128)
proj_qkv_kernel(const float* __restrict__ X, const unsigned short* __restrict__ Wb,
                const float* __restrict__ bias, unsigned short* __restrict__ out)
{
    const int lane = threadIdx.x & 31;
    const int wave = threadIdx.x >> 5;
    const int g    = lane >> 4;
    const int ln   = lane & 15;
    const int m0   = blockIdx.x * 16;
    const int n0   = blockIdx.y * 64 + wave * 16;

    v8f acc = {};
    for (int kk = 0; kk < HID_; kk += 32) {
        FragAB a, b;
        const float* arow = X + (size_t)(m0 + ln) * HID_ + kk;
        #pragma unroll
        for (int i = 0; i < 16; ++i) {
            int k = i + 16 * g;
            a.s[i] = f2bf(arow[k]);
        }
        const unsigned* wrow =
            (const unsigned*)(Wb + (size_t)(n0 + ln) * HID_ + kk + 16 * g);
        #pragma unroll
        for (int t = 0; t < 8; ++t) b.u[t] = wrow[t];
        acc = __builtin_amdgcn_wmma_f32_16x16x32_bf16(false, a.v, false, b.v,
                                                      (short)0, acc, false, false);
    }
    const int nn = n0 + ln;
    const int h  = nn >> 6;
    const int dk = nn & 63;
    const float bvl = bias[nn];
    #pragma unroll
    for (int r = 0; r < 8; ++r) {
        int mm = m0 + r + 8 * g;
        int bb = mm >> 12;
        int n  = mm & 4095;
        size_t idx = ((((size_t)bb * HEADS + h) * NSEQ) + n) * DK_ + dk;
        out[idx] = f2bf(acc[r] + bvl);
    }
}

// ---------------------------------------------------------------------------
// Kernel 2: S = 0.125 * Q @ K^T per (b,h).  Block computes a 64q x 64k tile:
// the shared 64x64 K tile arrives via one TDM load with LDS padding (33-dword
// row pitch -> conflict-free B reads); each wave computes 16q x 64k (8 WMMAs),
// reusing each Q fragment across 4 key sub-tiles.
// ---------------------------------------------------------------------------
#define KT_PITCH 66   // 64 bf16 + 2 pad (33 dwords)
__global__ void __launch_bounds__(128)
scores_kernel(const unsigned short* __restrict__ Qb,
              const unsigned short* __restrict__ Kb,
              float* __restrict__ attn)
{
    __shared__ unsigned short ktile[64 * KT_PITCH];   // 8.25 KB
    const int lane = threadIdx.x & 31;
    const int wave = threadIdx.x >> 5;
    const int g    = lane >> 4;
    const int ln   = lane & 15;
    const int q0   = blockIdx.x * 64 + wave * 16;
    const int k0   = blockIdx.y * 64;
    const size_t base = (size_t)blockIdx.z * NSEQ * DK_;

#if HAVE_TDM
    if (wave == 0) {
        // pad_interval=4: pad after every 32 dwords (one 64-elem bf16 row);
        // pad_amount=0: insert 1 dword -> row pitch 33 dwords.
        tdm_load_bf16_2d(0u, Kb + base + (size_t)k0 * DK_, DK_, 64, DK_, NSEQ,
                         1u, 4u, 0u);
        __builtin_amdgcn_s_wait_tensorcnt(0);
    }
    __syncthreads();
#else
    {
        unsigned* dst = (unsigned*)ktile;
        for (int i = threadIdx.x; i < 64 * 32; i += 128) {
            int r = i >> 5, c = i & 31;
            dst[r * 33 + c] =
                ((const unsigned*)(Kb + base + (size_t)(k0 + r) * DK_))[c];
        }
        __syncthreads();
    }
#endif

    const unsigned* qp = (const unsigned*)(Qb + base + (size_t)(q0 + ln) * DK_);
    v8f acc[4] = {{}, {}, {}, {}};
    #pragma unroll
    for (int dd = 0; dd < DK_; dd += 32) {
        FragAB a;
        #pragma unroll
        for (int t = 0; t < 8; ++t) {
            int d0 = dd + 2 * t + 16 * g;
            a.u[t] = qp[d0 >> 1];
        }
        #pragma unroll
        for (int nt = 0; nt < 4; ++nt) {
            FragAB b;
            const unsigned* kp =
                (const unsigned*)&ktile[(nt * 16 + ln) * KT_PITCH + dd + 16 * g];
            #pragma unroll
            for (int t = 0; t < 8; ++t) b.u[t] = kp[t];
            acc[nt] = __builtin_amdgcn_wmma_f32_16x16x32_bf16(
                          false, a.v, false, b.v, (short)0, acc[nt], false, false);
        }
    }
    float* arow = attn + (size_t)blockIdx.z * NSEQ * NSEQ;
    #pragma unroll
    for (int nt = 0; nt < 4; ++nt) {
        #pragma unroll
        for (int r = 0; r < 8; ++r) {
            int q = q0 + r + 8 * g;
            arow[(size_t)q * NSEQ + k0 + nt * 16 + ln] = acc[nt][r] * 0.125f;
        }
    }
}

// ---------------------------------------------------------------------------
// Kernel 3: row softmax over Nk=4096, in place (regs + LDS tree reduction).
// ---------------------------------------------------------------------------
__global__ void __launch_bounds__(256)
softmax_kernel(float* __restrict__ attn)
{
    const int tid = threadIdx.x;
    float* p = attn + (size_t)blockIdx.x * NSEQ;
    __shared__ float red[256];

    float e[16];
    float mx = -3.402823466e38f;
    #pragma unroll
    for (int i = 0; i < 16; ++i) { e[i] = p[tid + i * 256]; mx = fmaxf(mx, e[i]); }
    red[tid] = mx; __syncthreads();
    for (int s = 128; s > 0; s >>= 1) {
        if (tid < s) red[tid] = fmaxf(red[tid], red[tid + s]);
        __syncthreads();
    }
    mx = red[0]; __syncthreads();

    float sum = 0.f;
    #pragma unroll
    for (int i = 0; i < 16; ++i) { e[i] = __expf(e[i] - mx); sum += e[i]; }
    red[tid] = sum; __syncthreads();
    for (int s = 128; s > 0; s >>= 1) {
        if (tid < s) red[tid] += red[tid + s];
        __syncthreads();
    }
    float inv = 1.0f / red[0];
    #pragma unroll
    for (int i = 0; i < 16; ++i) p[tid + i * 256] = e[i] * inv;
}

// ---------------------------------------------------------------------------
// Kernel 4: Ctx = attn @ V.  V k-tiles staged in LDS via TDM (double-buffered,
// TENSORcnt-pipelined); 4 waves share the tile, each computes 16(q) x 64(d).
// ---------------------------------------------------------------------------
#define CTX_KT 32
__global__ void __launch_bounds__(128)
context_kernel(const float* __restrict__ attn,
               const unsigned short* __restrict__ Vb,
               unsigned short* __restrict__ Ctx)
{
    __shared__ unsigned short vtile[2][CTX_KT * DK_];   // 2 x 4 KB
    const int lane = threadIdx.x & 31;
    const int wave = threadIdx.x >> 5;
    const int g    = lane >> 4;
    const int ln   = lane & 15;
    const int q0   = blockIdx.x * 64 + wave * 16;
    const int bh   = blockIdx.z;

    const float* arow = attn + (size_t)bh * NSEQ * NSEQ + (size_t)(q0 + ln) * NSEQ;
    const unsigned short* vbase = Vb + (size_t)bh * NSEQ * DK_;

    v8f acc[4] = {{}, {}, {}, {}};

#if HAVE_TDM
    const unsigned tile_bytes = CTX_KT * DK_ * 2;
    if (wave == 0)
        tdm_load_bf16_2d(0u, vbase, DK_, CTX_KT, DK_, NSEQ);
#endif

    for (int kk = 0; kk < NSEQ; kk += CTX_KT) {
        const int cur = (kk / CTX_KT) & 1;
#if HAVE_TDM
        if (wave == 0) {
            if (kk + CTX_KT < NSEQ) {
                tdm_load_bf16_2d((unsigned)((cur ^ 1) * tile_bytes),
                                 vbase + (size_t)(kk + CTX_KT) * DK_,
                                 DK_, CTX_KT, DK_, NSEQ);
                __builtin_amdgcn_s_wait_tensorcnt(1);   // tile kk is resident
            } else {
                __builtin_amdgcn_s_wait_tensorcnt(0);
            }
        }
        __syncthreads();
#else
        {
            const unsigned* src = (const unsigned*)(vbase + (size_t)kk * DK_);
            unsigned* dst = (unsigned*)&vtile[cur][0];
            for (int i = threadIdx.x; i < CTX_KT * DK_ / 2; i += 128)
                dst[i] = src[i];
            __syncthreads();
        }
#endif
        FragAB a;
        #pragma unroll
        for (int i = 0; i < 16; ++i) {
            int k = kk + i + 16 * g;
            a.s[i] = f2bf(arow[k]);
        }
        #pragma unroll
        for (int dt = 0; dt < 4; ++dt) {
            FragAB b;
            #pragma unroll
            for (int i = 0; i < 16; ++i)
                b.s[i] = vtile[cur][(16 * g + i) * DK_ + dt * 16 + ln];
            acc[dt] = __builtin_amdgcn_wmma_f32_16x16x32_bf16(
                          false, a.v, false, b.v, (short)0, acc[dt], false, false);
        }
        __syncthreads();
    }

    const int bb = bh >> 3, h = bh & 7;
    #pragma unroll
    for (int dt = 0; dt < 4; ++dt) {
        #pragma unroll
        for (int r = 0; r < 8; ++r) {
            int q = q0 + r + 8 * g;
            Ctx[((size_t)bb * NSEQ + q) * HID_ + h * DK_ + dt * 16 + ln] =
                f2bf(acc[dt][r]);
        }
    }
}

// ---------------------------------------------------------------------------
// Kernel 5: out = Ctx @ Wo^T + bo  (bf16 A, bf16 W, fp32 out)
// ---------------------------------------------------------------------------
__global__ void __launch_bounds__(128)
outproj_kernel(const unsigned short* __restrict__ Ctx,
               const unsigned short* __restrict__ Wob,
               const float* __restrict__ bo,
               float* __restrict__ out)
{
    const int lane = threadIdx.x & 31;
    const int wave = threadIdx.x >> 5;
    const int g    = lane >> 4;
    const int ln   = lane & 15;
    const int m0   = blockIdx.x * 16;
    const int n0   = blockIdx.y * 64 + wave * 16;

    const unsigned* ap = (const unsigned*)(Ctx + (size_t)(m0 + ln) * HID_);

    v8f acc = {};
    for (int kk = 0; kk < HID_; kk += 32) {
        FragAB a, b;
        #pragma unroll
        for (int t = 0; t < 8; ++t) {
            int d0 = kk + 2 * t + 16 * g;
            a.u[t] = ap[d0 >> 1];
        }
        const unsigned* wrow =
            (const unsigned*)(Wob + (size_t)(n0 + ln) * HID_ + kk + 16 * g);
        #pragma unroll
        for (int t = 0; t < 8; ++t) b.u[t] = wrow[t];
        acc = __builtin_amdgcn_wmma_f32_16x16x32_bf16(false, a.v, false, b.v,
                                                      (short)0, acc, false, false);
    }
    const float bvl = bo[n0 + ln];
    #pragma unroll
    for (int r = 0; r < 8; ++r)
        out[(size_t)(m0 + r + 8 * g) * HID_ + n0 + ln] = acc[r] + bvl;
}

// ---------------------------------------------------------------------------
extern "C" void kernel_launch(void* const* d_in, const int* in_sizes, int n_in,
                              void* d_out, int out_size, void* d_ws, size_t ws_size,
                              hipStream_t stream)
{
    const float* query = (const float*)d_in[0];
    const float* key   = (const float*)d_in[1];
    const float* value = (const float*)d_in[2];
    const float* Wq    = (const float*)d_in[3];
    const float* bq    = (const float*)d_in[4];
    const float* Wk    = (const float*)d_in[5];
    const float* bk    = (const float*)d_in[6];
    const float* Wv    = (const float*)d_in[7];
    const float* bv    = (const float*)d_in[8];
    const float* Wo    = (const float*)d_in[9];
    const float* bo    = (const float*)d_in[10];

    const size_t qkv_elems = (size_t)BATCH * HEADS * NSEQ * DK_;  // 4,194,304
    const size_t w_elems   = (size_t)HID_ * HID_;                 // 262,144
    unsigned short* Qb  = (unsigned short*)d_ws;
    unsigned short* Kb  = Qb + qkv_elems;
    unsigned short* Vb  = Kb + qkv_elems;
    unsigned short* Ctx = Vb + qkv_elems;
    unsigned short* Wqb = Ctx + qkv_elems;
    unsigned short* Wkb = Wqb + w_elems;
    unsigned short* Wvb = Wkb + w_elems;
    unsigned short* Wob = Wvb + w_elems;

    float* out  = (float*)d_out;
    float* attn = out + (size_t)BN * HID_;   // tuple output #2

    dim3 cblk(256), cgrd(w_elems / 256);           // 1024 blocks
    wcvt_kernel<<<cgrd, cblk, 0, stream>>>(Wq, Wqb);
    wcvt_kernel<<<cgrd, cblk, 0, stream>>>(Wk, Wkb);
    wcvt_kernel<<<cgrd, cblk, 0, stream>>>(Wv, Wvb);
    wcvt_kernel<<<cgrd, cblk, 0, stream>>>(Wo, Wob);

    dim3 blk(128);
    dim3 g1(BN / 16, HID_ / 64);                   // (512, 8)
    proj_qkv_kernel<<<g1, blk, 0, stream>>>(query, Wqb, bq, Qb);
    proj_qkv_kernel<<<g1, blk, 0, stream>>>(key,   Wkb, bk, Kb);
    proj_qkv_kernel<<<g1, blk, 0, stream>>>(value, Wvb, bv, Vb);

    dim3 g2(NSEQ / 64, NSEQ / 64, BATCH * HEADS);  // (64, 64, 16)
    scores_kernel<<<g2, blk, 0, stream>>>(Qb, Kb, attn);

    softmax_kernel<<<dim3(BATCH * HEADS * NSEQ), dim3(256), 0, stream>>>(attn);

    dim3 g4(NSEQ / 64, 1, BATCH * HEADS);          // (64, 1, 16)
    context_kernel<<<g4, blk, 0, stream>>>(attn, Vb, Ctx);

    outproj_kernel<<<g1, blk, 0, stream>>>(Ctx, Wob, bo, out);
}
